// DNATransformer_1073741824689
// MI455X (gfx1250) — compile-verified, matrix operand order checked by reading, not run on hardware
//
#include <hip/hip_runtime.h>
#include <cstddef>
#include <cstdint>

#define Dm   1024
#define Sm   1024
#define Bm   2
#define Hm   16
#define HDm  64
#define Lm   6
#define Vm   4

typedef __attribute__((ext_vector_type(16))) __bf16 v16bf;
typedef __attribute__((ext_vector_type(8)))  __bf16 v8bf;
typedef __attribute__((ext_vector_type(8)))  float  v8f;

static __device__ __forceinline__ v8f wmma_bf16(v16bf a, v16bf b, v8f c) {
    return __builtin_amdgcn_wmma_f32_16x16x32_bf16(false, a, false, b, (short)0, c, false, false);
}

// load 8 consecutive f32 at p, convert to bf16, place at dst[off..off+7]
static __device__ __forceinline__ void cvt8(const float* p, v16bf& d, int off) {
    float4 a = *(const float4*)p;
    float4 b = *(const float4*)(p + 4);
    d[off + 0] = (__bf16)a.x; d[off + 1] = (__bf16)a.y;
    d[off + 2] = (__bf16)a.z; d[off + 3] = (__bf16)a.w;
    d[off + 4] = (__bf16)b.x; d[off + 5] = (__bf16)b.y;
    d[off + 6] = (__bf16)b.z; d[off + 7] = (__bf16)b.w;
}

// ---------------- embedding: x = tok[ids] + pos ----------------
__global__ void k_embed(const float* __restrict__ tok, const float* __restrict__ pos,
                        const int* __restrict__ ids, float* __restrict__ x) {
    int row = blockIdx.x;               // 0 .. B*S-1
    int s   = row % Sm;
    int id  = ids[row];
    const float* tr = tok + (size_t)id * Dm;
    const float* pr = pos + (size_t)s  * Dm;
    float* xr = x + (size_t)row * Dm;
    for (int i = threadIdx.x; i < Dm; i += blockDim.x) xr[i] = tr[i] + pr[i];
}

// ---------------- layernorm (one 256-thread block per token) ----------------
__global__ void k_ln(const float* __restrict__ x, const float* __restrict__ g,
                     const float* __restrict__ b, float* __restrict__ out) {
    int row = blockIdx.x;
    const float* xr = x + (size_t)row * Dm;
    float s = 0.f, s2 = 0.f;
    for (int i = threadIdx.x; i < Dm; i += blockDim.x) { float v = xr[i]; s += v; s2 += v * v; }
    __shared__ float r1[256], r2[256];
    r1[threadIdx.x] = s; r2[threadIdx.x] = s2;
    __syncthreads();
    for (int st = 128; st > 0; st >>= 1) {
        if ((int)threadIdx.x < st) { r1[threadIdx.x] += r1[threadIdx.x + st]; r2[threadIdx.x] += r2[threadIdx.x + st]; }
        __syncthreads();
    }
    float mean = r1[0] * (1.f / Dm);
    float var  = r2[0] * (1.f / Dm) - mean * mean;
    float rstd = rsqrtf(var + 1e-5f);
    float* orow = out + (size_t)row * Dm;
    for (int i = threadIdx.x; i < Dm; i += blockDim.x)
        orow[i] = (xr[i] - mean) * rstd * g[i] + b[i];
}

// ---------------- WMMA GEMM: C = act(A @ W + bias) [+ resid] ----------------
// A: MxK f32 row-major, W: KxN f32 row-major. Block 128 thr = 4 waves.
// Tile: 128 rows x 64 cols per block; 32 rows x 64 cols per wave (8 WMMA / K-step).
// FUSE_SILU: 0 = none, 1 = silu. resid == nullptr -> no residual add.
template <int FUSE_SILU>
__global__ __launch_bounds__(128) void k_gemm(const float* __restrict__ A,
                                              const float* __restrict__ W,
                                              const float* __restrict__ bias,
                                              const float* __restrict__ resid,
                                              float* __restrict__ C,
                                              int M, int N, int K) {
    __shared__ __align__(16) __bf16 wt[64 * 32];   // W tile transposed: [n][k], bf16
    const int tid  = threadIdx.x;
    const int wv   = tid >> 5;
    const int lane = tid & 31;
    const int hf   = lane >> 4;
    const int l16  = lane & 15;
    const int m0 = blockIdx.y * 128;
    const int n0 = blockIdx.x * 64;
    const int kb = hf * 8;                 // lane's first contraction run
    const float* ap0 = A + (size_t)(m0 + wv * 32 + l16) * K;
    const float* ap1 = A + (size_t)(m0 + wv * 32 + 16 + l16) * K;
    v8f acc[2][4] = {};
    for (int k0 = 0; k0 < K; k0 += 32) {
        // stage W[k0..k0+31][n0..n0+63] -> wt[n*32+k] (coalesced reads, transposed bf16 writes)
        #pragma unroll
        for (int i = 0; i < 16; i++) {
            int idx = i * 128 + tid;
            int kk = idx >> 6, nn = idx & 63;
            int n = n0 + nn;
            float v = (n < N) ? W[(size_t)(k0 + kk) * N + n] : 0.f;
            wt[nn * 32 + kk] = (__bf16)v;
        }
        if (k0 + 32 < K) __builtin_prefetch(W + (size_t)(k0 + 32) * N + n0 + (tid & 63), 0, 1);
        __syncthreads();
        // A fragments: lane(lo): K = 0..7,16..23 ; lane(hi): K = 8..15,24..31
        v16bf a0, a1;
        cvt8(ap0 + k0 + kb,      a0, 0);
        cvt8(ap0 + k0 + kb + 16, a0, 8);
        cvt8(ap1 + k0 + kb,      a1, 0);
        cvt8(ap1 + k0 + kb + 16, a1, 8);
        #pragma unroll
        for (int t = 0; t < 4; t++) {
            const __bf16* wp = &wt[(t * 16 + l16) * 32 + kb];
            v8bf lo = *(const v8bf*)wp;
            v8bf hi = *(const v8bf*)(wp + 16);
            v16bf bv = __builtin_shufflevector(lo, hi, 0, 1, 2, 3, 4, 5, 6, 7,
                                               8, 9, 10, 11, 12, 13, 14, 15);
            acc[0][t] = wmma_bf16(a0, bv, acc[0][t]);
            acc[1][t] = wmma_bf16(a1, bv, acc[1][t]);
        }
        __syncthreads();
    }
    // epilogue: C layout -> element (v, lane): row = base + v + 8*hf, col = n0+t*16+l16
    #pragma unroll
    for (int rr = 0; rr < 2; rr++) {
        #pragma unroll
        for (int t = 0; t < 4; t++) {
            int col = n0 + t * 16 + l16;
            if (col >= N) continue;
            float bb = bias ? bias[col] : 0.f;
            #pragma unroll
            for (int v = 0; v < 8; v++) {
                int r = m0 + wv * 32 + rr * 16 + v + 8 * hf;
                float val = acc[rr][t][v] + bb;
                if (FUSE_SILU) val = val / (1.f + __expf(-val));   // silu
                if (resid) val += resid[(size_t)r * N + col];
                C[(size_t)r * N + col] = val;
            }
        }
    }
}

// ---------------- flash attention (transposed): qkv (B,S,H,3*HD) -> av (B,S,D) ----
// Computes S^T = K·Q^T and O^T = V^T·P^T so that each query lives in one lane:
// softmax stats are per-lane scalars (one shfl_xor(16) per 32-key block) and the
// exponentiated scores are already in P^T's B-fragment layout (no LDS relay).
// grid (S/64, H, B), block 128 = 4 waves; wave handles 16 query rows.
__global__ __launch_bounds__(128) void k_attn(const float* __restrict__ qkv,
                                              float* __restrict__ av) {
    const int lane = threadIdx.x & 31;
    const int wv   = threadIdx.x >> 5;
    const int hf   = lane >> 4;
    const int l16  = lane & 15;
    const int h  = blockIdx.y;
    const int bb = blockIdx.z;
    const int q0 = blockIdx.x * 64 + wv * 16;
    const float scl = 0.125f;                     // 1/sqrt(HD)
    const size_t srs = (size_t)Hm * 3 * HDm;      // stride between sequence positions
    const float* base = qkv + ((size_t)bb * Sm * Hm + h) * (3 * HDm);
    const int kb = hf * 8;
    const int qcol = q0 + l16;                    // this lane's query

    // B-fragment: Q^T (lane = query column, elements = hd contraction runs)
    const float* qrow = base + (size_t)qcol * srs;
    v16bf bq[2];
    cvt8(qrow + kb,           bq[0], 0);
    cvt8(qrow + kb + 16,      bq[0], 8);
    cvt8(qrow + 32 + kb,      bq[1], 0);
    cvt8(qrow + 32 + kb + 16, bq[1], 8);

    v8f o[4] = {};                                // O^T: row = hd (per 16-tile), col = query
    float mrun = -1e30f, lrun = 0.f;

    for (int j0 = 0; j0 <= q0 + 15; j0 += 32) {
        v8f st[2];
        // transposed scores, two 16-key sub-tiles: element (v,lane) = (key v+8hf, query l16)
        #pragma unroll
        for (int kt = 0; kt < 2; kt++) {
            int key = j0 + kt * 16 + l16;         // A-fragment row = key
            v16bf ak0, ak1;
            if (key < Sm) {
                const float* krow = base + (size_t)key * srs + HDm;   // K block
                cvt8(krow + kb,           ak0, 0);
                cvt8(krow + kb + 16,      ak0, 8);
                cvt8(krow + 32 + kb,      ak1, 0);
                cvt8(krow + 32 + kb + 16, ak1, 8);
            } else {
                #pragma unroll
                for (int j = 0; j < 16; j++) { ak0[j] = (__bf16)0.f; ak1[j] = (__bf16)0.f; }
            }
            v8f s = {};
            s = wmma_bf16(ak0, bq[0], s);
            s = wmma_bf16(ak1, bq[1], s);
            #pragma unroll
            for (int v = 0; v < 8; v++) {
                int keyv = j0 + kt * 16 + v + 8 * hf;
                float sv = s[v] * scl;
                if (keyv > qcol || keyv >= Sm) sv = -1e30f;   // causal mask
                s[v] = sv;
            }
            st[kt] = s;
        }
        // per-query max over the 32 keys: 16 in-lane values + one cross-half shuffle
        float tmax = -1e30f;
        #pragma unroll
        for (int v = 0; v < 8; v++) tmax = fmaxf(tmax, fmaxf(st[0][v], st[1][v]));
        tmax = fmaxf(tmax, __shfl_xor(tmax, 16, 32));
        float mnew = fmaxf(mrun, tmax);
        float rsc  = __expf(mrun - mnew);
        mrun = mnew;
        // exponentiate; bp IS the B-fragment of P^T (lane=query, elements=key runs)
        v16bf bp;
        float psum = 0.f;
        #pragma unroll
        for (int v = 0; v < 8; v++) {
            float p0 = __expf(st[0][v] - mnew);   // key  8*hf + v        (tile 0)
            float p1 = __expf(st[1][v] - mnew);   // key  16 + 8*hf + v   (tile 1)
            psum += p0 + p1;
            bp[v]     = (__bf16)p0;
            bp[v + 8] = (__bf16)p1;
        }
        psum += __shfl_xor(psum, 16, 32);
        lrun = lrun * rsc + psum;
        // rescale O^T (per-lane scalar) and accumulate V^T @ P^T
        #pragma unroll
        for (int t = 0; t < 4; t++) {
            #pragma unroll
            for (int v = 0; v < 8; v++) o[t][v] *= rsc;
        }
        #pragma unroll
        for (int t = 0; t < 4; t++) {
            // A-fragment: V^T tile (row = hd t*16+l16, elements = key runs)
            v16bf avt;
            int hd = t * 16 + l16;
            #pragma unroll
            for (int j = 0; j < 8; j++) {
                int k1 = j0 + kb + j;
                int k2 = k1 + 16;
                float v1 = (k1 < Sm) ? base[(size_t)k1 * srs + 2 * HDm + hd] : 0.f;
                float v2 = (k2 < Sm) ? base[(size_t)k2 * srs + 2 * HDm + hd] : 0.f;
                avt[j]     = (__bf16)v1;
                avt[j + 8] = (__bf16)v2;
            }
            o[t] = wmma_bf16(avt, bp, o[t]);
        }
    }

    // normalize & store: element (t,v): hd = t*16 + v + 8*hf, query = qcol (contiguous in v)
    float inv = 1.f / lrun;
    float* orow = av + ((size_t)bb * Sm + qcol) * Dm + h * HDm;
    #pragma unroll
    for (int t = 0; t < 4; t++) {
        float4 lo4 = make_float4(o[t][0] * inv, o[t][1] * inv, o[t][2] * inv, o[t][3] * inv);
        float4 hi4 = make_float4(o[t][4] * inv, o[t][5] * inv, o[t][6] * inv, o[t][7] * inv);
        float* p = orow + t * 16 + 8 * hf;
        *(float4*)p       = lo4;
        *(float4*)(p + 4) = hi4;
    }
}

// ---------------- finalize: preds = logits[:, :-1, :], labels = ids[:, 1:] ----------------
__global__ void k_final(const float* __restrict__ logits, const int* __restrict__ ids,
                        float* __restrict__ outF, int* __restrict__ outI) {
    int i = blockIdx.x * blockDim.x + threadIdx.x;
    if (i >= Bm * (Sm - 1)) return;
    int b = i / (Sm - 1), s = i % (Sm - 1);
    const float* lr = logits + ((size_t)(b * Sm + s)) * Vm;
    float* pr = outF + (size_t)i * Vm;
    #pragma unroll
    for (int v = 0; v < Vm; v++) pr[v] = lr[v];
    outI[i] = ids[b * Sm + s + 1];
}

extern "C" void kernel_launch(void* const* d_in, const int* in_sizes, int n_in,
                              void* d_out, int out_size, void* d_ws, size_t ws_size,
                              hipStream_t stream) {
    // setup_inputs() dict order (params flattened in insertion order, then input_ids)
    const float* tok    = (const float*)d_in[0];   // (V, D)
    const float* pos    = (const float*)d_in[1];   // (S, D)
    const float* ln1_g  = (const float*)d_in[2];   // (L, D)
    const float* ln1_b  = (const float*)d_in[3];
    const float* qkv_w  = (const float*)d_in[4];   // (L, D, 3D)
    const float* qkv_b  = (const float*)d_in[5];   // (L, 3D)
    const float* out_w  = (const float*)d_in[6];   // (L, D, D)
    const float* out_b  = (const float*)d_in[7];   // (L, D)
    const float* ln2_g  = (const float*)d_in[8];
    const float* ln2_b  = (const float*)d_in[9];
    const float* fc1_w  = (const float*)d_in[10];  // (L, D, 4D)
    const float* fc1_b  = (const float*)d_in[11];  // (L, 4D)
    const float* fc2_w  = (const float*)d_in[12];  // (L, 4D, D)
    const float* fc2_b  = (const float*)d_in[13];  // (L, D)
    const float* hln_g  = (const float*)d_in[14];  // (D)
    const float* hln_b  = (const float*)d_in[15];
    const float* head_w = (const float*)d_in[16];  // (D, V)
    const float* head_b = (const float*)d_in[17];  // (V)
    const int*   ids    = (const int*)d_in[18];    // (B, S)

    const int M = Bm * Sm;                         // 2048 tokens
    float* ws = (float*)d_ws;
    float* x  = ws;                                // (M, D)
    float* hb = x  + (size_t)M * Dm;               // (M, D)   LN output
    float* qk = hb + (size_t)M * Dm;               // (M, 3D)  qkv
    float* avb= qk + (size_t)M * 3 * Dm;           // (M, D)   attention output
    float* ff = avb+ (size_t)M * Dm;               // (M, 4D)  ffn hidden
    float* lg = ff + (size_t)M * 4 * Dm;           // (M, V)   logits

    k_embed<<<M, 256, 0, stream>>>(tok, pos, ids, x);

    for (int l = 0; l < Lm; l++) {
        size_t lD = (size_t)l * Dm;
        k_ln<<<M, 256, 0, stream>>>(x, ln1_g + lD, ln1_b + lD, hb);

        dim3 gq((3 * Dm) / 64, M / 128);
        k_gemm<0><<<gq, 128, 0, stream>>>(hb, qkv_w + (size_t)l * Dm * 3 * Dm,
                                          qkv_b + (size_t)l * 3 * Dm, nullptr, qk,
                                          M, 3 * Dm, Dm);

        dim3 ga(Sm / 64, Hm, Bm);
        k_attn<<<ga, 128, 0, stream>>>(qk, avb);

        dim3 go(Dm / 64, M / 128);
        k_gemm<0><<<go, 128, 0, stream>>>(avb, out_w + (size_t)l * Dm * Dm,
                                          out_b + lD, x, x, M, Dm, Dm);

        k_ln<<<M, 256, 0, stream>>>(x, ln2_g + lD, ln2_b + lD, hb);

        dim3 g1((4 * Dm) / 64, M / 128);
        k_gemm<1><<<g1, 128, 0, stream>>>(hb, fc1_w + (size_t)l * Dm * 4 * Dm,
                                          fc1_b + (size_t)l * 4 * Dm, nullptr, ff,
                                          M, 4 * Dm, Dm);

        dim3 g2(Dm / 64, M / 128);
        k_gemm<0><<<g2, 128, 0, stream>>>(ff, fc2_w + (size_t)l * 4 * Dm * Dm,
                                          fc2_b + lD, x, x, M, Dm, 4 * Dm);
    }

    k_ln<<<M, 256, 0, stream>>>(x, hln_g, hln_b, hb);
    dim3 gh(1, M / 128);                           // N = V = 4 (guarded inside)
    k_gemm<0><<<gh, 128, 0, stream>>>(hb, head_w, head_b, nullptr, lg, M, Vm, Dm);

    int npl = Bm * (Sm - 1);
    float* outF = (float*)d_out;
    int*   outI = (int*)(outF + (size_t)npl * Vm);
    k_final<<<(npl + 255) / 256, 256, 0, stream>>>(lg, ids, outF, outI);
}
